// MultiHeadAttention_90537910599844
// MI455X (gfx1250) — matrix operand order repaired and checked by reading, not compiled, not run on hardware
//
#include <hip/hip_runtime.h>
#include <hip/hip_bf16.h>

// ---------------------------------------------------------------------------
// MI455X (gfx1250) multi-head attention: QKV GEMM -> flash attention -> proj
// All matmuls via v_wmma_f32_16x16x32_bf16 (fp32 accumulate, bf16 operands).
// Softmax row-sums computed inside the WMMA pipe via a ones-column accumulator.
// Next-tile global_prefetch_b8 to hide HBM/L2 latency across barriers.
// ---------------------------------------------------------------------------

typedef __attribute__((ext_vector_type(16))) __bf16 v16bf;
typedef __attribute__((ext_vector_type(8)))  float  v8f;

union Frag {
    v16bf v;
    unsigned int u[8];
};

__device__ __forceinline__ unsigned short f2bf(float f) {
    unsigned int u = __float_as_uint(f);
    unsigned int r = u + 0x7FFFu + ((u >> 16) & 1u);  // round-to-nearest-even
    return (unsigned short)(r >> 16);
}

// ---------------------------------------------------------------------------
// GEMM: C[M,N] = A[M,K] * W[K,N] + bias[N]
//   ABF16: A is bf16 (else fp32, converted on load)
//   CBF16: C stored as bf16 (else fp32)
// Block tile 128(M) x 64(N), 8 waves as 4x2, each wave 32x32 (2x2 WMMA tiles),
// K staged 32-wide through LDS; next K-slice prefetched while current loads.
// ---------------------------------------------------------------------------
template <int ABF16, int CBF16>
__global__ __launch_bounds__(256) void gemm_bias_kernel(
    const void* __restrict__ Ap, const float* __restrict__ W,
    const float* __restrict__ bias, void* __restrict__ Cp,
    int M, int N, int K) {
    __shared__ unsigned short As[128][32];  // A tile, row-major (bf16 bits)
    __shared__ unsigned short Wt[64][32];   // W tile transposed: [n][k]

    const int tid  = threadIdx.x;
    const int wid  = tid >> 5;          // wave id 0..7 (wave32)
    const int lane = tid & 31;
    const int hi   = lane >> 4;         // half-wave select
    const int l16  = lane & 15;
    const int wm   = wid >> 1;          // 0..3
    const int wn   = wid & 1;           // 0..1
    const int m0   = wm * 32;
    const int n0   = wn * 32;
    const int gm   = blockIdx.y * 128;
    const int gn   = blockIdx.x * 64;

    v8f acc[2][2];
#pragma unroll
    for (int i = 0; i < 2; ++i)
#pragma unroll
        for (int j = 0; j < 2; ++j)
#pragma unroll
            for (int e = 0; e < 8; ++e) acc[i][j][e] = 0.0f;

    for (int kt = 0; kt < K; kt += 32) {
        const bool more = (kt + 32) < K;
        // ---- cooperative load A tile (128x32), prefetch next slice ----
#pragma unroll
        for (int p = 0; p < 16; ++p) {
            int idx = tid + p * 256;
            int r = idx >> 5, c = idx & 31;
            unsigned short v;
            if (ABF16) {
                const unsigned short* a = (const unsigned short*)Ap;
                v = a[(size_t)(gm + r) * K + kt + c];
                if (more) __builtin_prefetch(&a[(size_t)(gm + r) * K + kt + 32 + c], 0, 3);
            } else {
                const float* a = (const float*)Ap;
                v = f2bf(a[(size_t)(gm + r) * K + kt + c]);
                if (more) __builtin_prefetch(&a[(size_t)(gm + r) * K + kt + 32 + c], 0, 3);
            }
            As[r][c] = v;
        }
        // ---- cooperative load W tile (32x64) transposed, prefetch next ----
#pragma unroll
        for (int p = 0; p < 8; ++p) {
            int idx = tid + p * 256;
            int r = idx >> 6, c = idx & 63;
            Wt[c][r] = f2bf(W[(size_t)(kt + r) * N + gn + c]);
            if (more) __builtin_prefetch(&W[(size_t)(kt + 32 + r) * N + gn + c], 0, 3);
        }
        __syncthreads();

        // ---- gather fragments per ISA 16-bit layouts ----
        Frag a[2], b[2];
#pragma unroll
        for (int i = 0; i < 2; ++i)
#pragma unroll
            for (int j = 0; j < 8; ++j) {
                int kk = 2 * (j & 3) + 16 * (j >> 2) + 8 * hi;  // A layout
                a[i].u[j] = *(const unsigned int*)&As[m0 + i * 16 + l16][kk];
            }
#pragma unroll
        for (int i = 0; i < 2; ++i)
#pragma unroll
            for (int j = 0; j < 8; ++j) {
                int kk = 2 * (j & 3) + 16 * (j >> 2) + 8 * hi;  // B layout
                b[i].u[j] = *(const unsigned int*)&Wt[n0 + i * 16 + l16][kk];
            }
#pragma unroll
        for (int i = 0; i < 2; ++i)
#pragma unroll
            for (int j = 0; j < 2; ++j)
                acc[i][j] = __builtin_amdgcn_wmma_f32_16x16x32_bf16(
                    false, a[i].v, false, b[j].v, (short)0, acc[i][j], false, false);
        __syncthreads();
    }

    // ---- store: D layout M = r + 8*hi, N = lane&15 ----
#pragma unroll
    for (int i = 0; i < 2; ++i)
#pragma unroll
        for (int j = 0; j < 2; ++j)
#pragma unroll
            for (int r = 0; r < 8; ++r) {
                int row = gm + m0 + i * 16 + r + 8 * hi;
                int col = gn + n0 + j * 16 + l16;
                float v = acc[i][j][r] + bias[col];
                if (CBF16)
                    ((unsigned short*)Cp)[(size_t)row * N + col] = f2bf(v);
                else
                    ((float*)Cp)[(size_t)row * N + col] = v;
            }
}

// ---------------------------------------------------------------------------
// Flash attention with the reference's MULTIPLICATIVE mask:
//   s' = (q.k / 8) * (mask * -1e9)    (mask==0 -> s'==0, mask==1 -> s*-1e9)
// Block: one (b,h) pair, 128 q-rows; 8 waves x 16 q-rows each.
// Key tiles of 32; online softmax; row sums via P @ ones-column WMMA.
// qkv layout: bf16 [B*S, 3072] (q at +0, k at +1024, v at +2048, head-packed).
// ---------------------------------------------------------------------------
__global__ __launch_bounds__(256) void attn_kernel(
    const unsigned short* __restrict__ qkv, const float* __restrict__ mask,
    unsigned short* __restrict__ attn_out) {
    const int S = 1024, H = 16, DH = 64, TD = 3072, D = 1024;

    const int b  = blockIdx.x / H;
    const int h  = blockIdx.x % H;
    const int q0 = blockIdx.y * 128;

    const int tid  = threadIdx.x;
    const int w    = tid >> 5;
    const int lane = tid & 31;
    const int hi   = lane >> 4;
    const int l16  = lane & 15;

    __shared__ unsigned short Qs[128][64];      // q tile (bf16)
    __shared__ unsigned short Ks[32][64];       // k tile [key][dh]
    __shared__ unsigned short Vst[64][32];      // v tile transposed [dh][key]
    __shared__ float          Ms[128][32];      // mask tile [q][key]
    __shared__ unsigned short Ps[8][16][32];    // per-wave P transpose buffer

    // ---- load Q tile (already bf16 in workspace) ----
#pragma unroll
    for (int p = 0; p < 32; ++p) {
        int idx = tid + p * 256;
        int r = idx >> 6, c = idx & 63;
        Qs[r][c] = qkv[(size_t)(b * S + q0 + r) * TD + h * DH + c];
    }
    __syncthreads();

    // q A-fragments: rows w*16..w*16+15, K = dh split into two 32-chunks
    Frag qa[2];
#pragma unroll
    for (int f = 0; f < 2; ++f)
#pragma unroll
        for (int j = 0; j < 8; ++j) {
            int kk = f * 32 + 2 * (j & 3) + 16 * (j >> 2) + 8 * hi;
            qa[f].u[j] = *(const unsigned int*)&Qs[w * 16 + l16][kk];
        }

    // ones-column B fragment: column N==0 is bf16 1.0, all other columns 0.
    // P @ ones accumulates per-row softmax denominators in the WMMA pipe.
    Frag ones;
#pragma unroll
    for (int j = 0; j < 8; ++j) ones.u[j] = (l16 == 0) ? 0x3F803F80u : 0u;

    float mrow[8];
    v8f o[5];  // 4 dh tiles + 1 row-sum (denominator) tile
#pragma unroll
    for (int r = 0; r < 8; ++r) mrow[r] = -3.0e38f;
#pragma unroll
    for (int t = 0; t < 5; ++t)
#pragma unroll
        for (int e = 0; e < 8; ++e) o[t][e] = 0.0f;

    for (int kb = 0; kb < 32; ++kb) {
        const int key0 = kb * 32;
        const bool more = kb < 31;
        __syncthreads();  // protect previous tiles until all waves done
        // ---- K and V tiles (32 keys x 64 dh), V transposed; prefetch next ----
#pragma unroll
        for (int p = 0; p < 8; ++p) {
            int idx = tid + p * 256;
            int r = idx >> 6, c = idx & 63;
            size_t base = (size_t)(b * S + key0 + r) * TD + h * DH + c;
            Ks[r][c]  = qkv[base + 1024];
            Vst[c][r] = qkv[base + 2048];
            if (more) {
                __builtin_prefetch(&qkv[base + (size_t)32 * TD + 1024], 0, 3);
                __builtin_prefetch(&qkv[base + (size_t)32 * TD + 2048], 0, 3);
            }
        }
        // ---- mask tile (128 q x 32 key) ----
#pragma unroll
        for (int p = 0; p < 16; ++p) {
            int idx = tid + p * 256;
            int r = idx >> 5, c = idx & 31;
            size_t mbase = (size_t)b * S * S + (size_t)(q0 + r) * S + key0 + c;
            Ms[r][c] = mask[mbase];
            if (more) __builtin_prefetch(&mask[mbase + 32], 0, 3);
        }
        __syncthreads();

        // ---- scores: 16(q) x 32(key), K-dim = dh = 64 (2 chained WMMA) ----
        v8f sc[2];
#pragma unroll
        for (int sub = 0; sub < 2; ++sub) {
            Frag kf0, kf1;
#pragma unroll
            for (int j = 0; j < 8; ++j) {
                kf0.u[j] = *(const unsigned int*)&Ks[sub * 16 + l16][2 * (j & 3) + 16 * (j >> 2) + 8 * hi];
                kf1.u[j] = *(const unsigned int*)&Ks[sub * 16 + l16][32 + 2 * (j & 3) + 16 * (j >> 2) + 8 * hi];
            }
            v8f z;
#pragma unroll
            for (int e = 0; e < 8; ++e) z[e] = 0.0f;
            z = __builtin_amdgcn_wmma_f32_16x16x32_bf16(false, qa[0].v, false, kf0.v,
                                                        (short)0, z, false, false);
            z = __builtin_amdgcn_wmma_f32_16x16x32_bf16(false, qa[1].v, false, kf1.v,
                                                        (short)0, z, false, false);
            sc[sub] = z;
        }

        // ---- multiplicative mask + online softmax (max reduction only) ----
#pragma unroll
        for (int r = 0; r < 8; ++r) {
            int rq = w * 16 + r + 8 * hi;  // local q row of this score element
            float s0 = sc[0][r] * 0.125f * (Ms[rq][l16] * -1e9f);
            float s1 = sc[1][r] * 0.125f * (Ms[rq][16 + l16] * -1e9f);
            float tm = fmaxf(s0, s1);
#pragma unroll
            for (int d = 1; d < 16; d <<= 1) tm = fmaxf(tm, __shfl_xor(tm, d, 32));
            float mn   = fmaxf(mrow[r], tm);
            float corr = __expf(mrow[r] - mn);
            mrow[r] = mn;
            s0 = __expf(s0 - mn);
            s1 = __expf(s1 - mn);
#pragma unroll
            for (int t = 0; t < 5; ++t) o[t][r] *= corr;
            // transpose P into A-fragment layout via per-wave LDS
            Ps[w][r + 8 * hi][l16]      = f2bf(s0);
            Ps[w][r + 8 * hi][16 + l16] = f2bf(s1);
        }
        __syncthreads();

        Frag pa;
#pragma unroll
        for (int j = 0; j < 8; ++j) {
            int kk = 2 * (j & 3) + 16 * (j >> 2) + 8 * hi;
            pa.u[j] = *(const unsigned int*)&Ps[w][l16][kk];
        }
        // ---- O += P(16x32) @ V(32x64): 4 WMMA + 1 denominator WMMA ----
#pragma unroll
        for (int t = 0; t < 4; ++t) {
            Frag vb;
#pragma unroll
            for (int j = 0; j < 8; ++j)
                vb.u[j] = *(const unsigned int*)&Vst[t * 16 + l16][2 * (j & 3) + 16 * (j >> 2) + 8 * hi];
            o[t] = __builtin_amdgcn_wmma_f32_16x16x32_bf16(false, pa.v, false, vb.v,
                                                           (short)0, o[t], false, false);
        }
        o[4] = __builtin_amdgcn_wmma_f32_16x16x32_bf16(false, pa.v, false, ones.v,
                                                       (short)0, o[4], false, false);
    }

    // ---- normalize and store attention output (bf16, head-merged [B*S, D]) ----
#pragma unroll
    for (int r = 0; r < 8; ++r) {
        // row-sum for row (r + 8*hi) lives in lane (16*hi), element r
        float l   = __shfl(o[4][r], lane & 16, 32);
        float inv = 1.0f / l;
        int grow = q0 + w * 16 + r + 8 * hi;
#pragma unroll
        for (int t = 0; t < 4; ++t) {
            int col = h * DH + t * 16 + l16;
            attn_out[(size_t)(b * S + grow) * D + col] = f2bf(o[t][r] * inv);
        }
    }
}

// ---------------------------------------------------------------------------
// Launch: gemm(QKV, fp32->bf16) -> attention -> gemm(proj, bf16->fp32)
// Workspace: qkv bf16 (48 MiB) + attn bf16 (16 MiB) = 64 MiB.
// ---------------------------------------------------------------------------
extern "C" void kernel_launch(void* const* d_in, const int* in_sizes, int n_in,
                              void* d_out, int out_size, void* d_ws, size_t ws_size,
                              hipStream_t stream) {
    (void)in_sizes; (void)n_in; (void)out_size; (void)ws_size;

    const float* x      = (const float*)d_in[0];
    const float* mask   = (const float*)d_in[1];
    const float* W_attn = (const float*)d_in[2];
    const float* b_attn = (const float*)d_in[3];
    const float* W_proj = (const float*)d_in[4];
    const float* b_proj = (const float*)d_in[5];
    float* out = (float*)d_out;

    const int M = 8 * 1024;   // B*S
    const int Dm = 1024;
    unsigned short* qkv  = (unsigned short*)d_ws;
    unsigned short* attn = qkv + (size_t)M * 3 * Dm;

    // 1) QKV projection: [8192,1024] x [1024,3072] -> bf16 qkv
    gemm_bias_kernel<0, 1><<<dim3(3 * Dm / 64, M / 128), 256, 0, stream>>>(
        x, W_attn, b_attn, qkv, M, 3 * Dm, Dm);

    // 2) Flash attention per (b,h), 128 q-rows per block
    attn_kernel<<<dim3(8 * 16, 1024 / 128), 256, 0, stream>>>(qkv, mask, attn);

    // 3) Output projection: [8192,1024] x [1024,1024] -> fp32 out
    gemm_bias_kernel<1, 0><<<dim3(Dm / 64, M / 128), 256, 0, stream>>>(
        attn, W_proj, b_proj, out, M, Dm, Dm);
}
